// ProposalNetwork_61314953118119
// MI455X (gfx1250) — matrix operand order repaired and checked by reading, not compiled, not run on hardware
//
#include <hip/hip_runtime.h>
#include <hip/hip_bf16.h>

typedef __attribute__((ext_vector_type(16))) _Float16 v16h;
typedef __attribute__((ext_vector_type(8)))  _Float16 v8h;
typedef __attribute__((ext_vector_type(8)))  float    v8f;

#define ROWS 128                 // rows (points) per block
#define F    256                 // hidden width
#define AS   264                 // padded LDS stride in halves (16B-aligned rows)
#define WS   264
#define WIMG_HALVES (F * WS)     // 67584 halves per weight image
#define WIMG_BYTES  (WIMG_HALVES * 2)  // 135168 B

// Gaussian activation: exp(-x^2 * inv2s2), inv2s2 = 1/(2*sigma^2) precomputed
__device__ __forceinline__ float gaussf(float x, float inv2s2) {
    return __expf(-x * x * inv2s2);
}

// A fragment: 16x32 f16 tile of activations (row-major [m][k] in LDS).
//   lane l=lane&15 -> M=m0+l; hi=lane>>4: V0-3 K=k0+hi*8+{0..7}, V4-7 K=k0+16+hi*8+{0..7}
__device__ __forceinline__ v16h load_a_frag(const _Float16* src, int m0, int k0, int l, int hi) {
    const _Float16* p = src + (m0 + l) * AS + k0 + hi * 8;
    v8h lo = *(const v8h*)(p);        // ds_load_b128
    v8h h8 = *(const v8h*)(p + 16);   // ds_load_b128
    v16h a;
#pragma unroll
    for (int i = 0; i < 8; ++i) { a[i] = lo[i]; a[8 + i] = h8[i]; }
    return a;
}

// B fragment: 32x16 f16 tile from TRANSPOSED weights Wt[n][k]:
//   lane: N = n0 + (lane&15); V0-3 K = k0 + (lane>>4)*8 + {0..7}, V4-7 K = k0 + 16 + (lane>>4)*8 + {0..7}
__device__ __forceinline__ v16h load_b_frag(const _Float16* wt, int k0, int n0, int l, int hi) {
    const _Float16* p = wt + (n0 + l) * WS + k0 + hi * 8;
    v8h lo = *(const v8h*)(p);        // ds_load_b128
    v8h h8 = *(const v8h*)(p + 16);   // ds_load_b128
    v16h b;
#pragma unroll
    for (int i = 0; i < 8; ++i) { b[i] = lo[i]; b[8 + i] = h8[i]; }
    return b;
}

// Straight B128 copy of a pre-converted f16 weight image (exact LDS layout) + tables.
__device__ __forceinline__ void stage_from_img(const _Float16* __restrict__ img,
                                               const float* __restrict__ tbl,
                                               _Float16* wt, float* bbuf, float* ibuf,
                                               int tid) {
    bbuf[tid] = tbl[tid];        // bias
    ibuf[tid] = tbl[F + tid];    // 1/(2*sigma^2)
    const uint4* s = (const uint4*)img;
    uint4* d = (uint4*)wt;
    const int n128 = WIMG_BYTES / 16;   // 8448 chunks
    for (int i = tid; i < n128; i += 256) d[i] = s[i];   // global_load_b128 -> ds_store_b128
}

// Warm the cache for the next weight image while VALU work runs (global_prefetch_b8).
__device__ __forceinline__ void prefetch_img(const _Float16* img, int tid) {
    const char* p = (const char*)img;
    for (int off = tid * 128; off < WIMG_BYTES; off += 256 * 128)
        __builtin_prefetch(p + off, 0, 1);
}

// One 256->256 layer: dst = gauss(src @ W + b, sigma).
// Each wave owns a 32-row x 128-col macro tile (2 M-tiles x 8 N-tiles), so every
// B fragment read from LDS feeds TWO WMMAs -> half the DS bandwidth of M=16 tiling,
// and two independent accumulator chains for the XDL pipe.
__device__ __forceinline__ void wmma_layer(const _Float16* src, _Float16* dst,
                                           const _Float16* wt,
                                           const float* bbuf, const float* ibuf,
                                           int wave, int l, int hi) {
    const int m0 = (wave & 3) * 32;        // rows [m0, m0+32)
    const int jb = (wave >> 2) * 8;        // col tiles [jb, jb+8)

    v16h a0[8], a1[8];                     // 16 A-fragments resident (128 VGPRs)
#pragma unroll
    for (int k = 0; k < 8; ++k) {
        a0[k] = load_a_frag(src, m0,      k * 32, l, hi);
        a1[k] = load_a_frag(src, m0 + 16, k * 32, l, hi);
    }

    for (int jj = 0; jj < 8; ++jj) {
        const int n0 = (jb + jj) * 16;
        float bv  = bbuf[n0 + l];   // ds_load_2addr pair with ibuf
        float inv = ibuf[n0 + l];

        v8f acc0 = {}, acc1 = {};
#pragma unroll
        for (int k = 0; k < 8; ++k) {
            v16h bf = load_b_frag(wt, k * 32, n0, l, hi);
            acc0 = __builtin_amdgcn_wmma_f32_16x16x32_f16(
                false, a0[k], false, bf, (short)0, acc0, false, false);
            acc1 = __builtin_amdgcn_wmma_f32_16x16x32_f16(
                false, a1[k], false, bf, (short)0, acc1, false, false);
        }
        // C/D layout: VGPR r holds element (M = r + hi*8, N = l)
#pragma unroll
        for (int r = 0; r < 8; ++r) {
            float x0 = acc0[r] + bv;
            dst[(m0 + r + hi * 8) * AS + n0 + l]      = (_Float16)gaussf(x0, inv);
            float x1 = acc1[r] + bv;
            dst[(m0 + 16 + r + hi * 8) * AS + n0 + l] = (_Float16)gaussf(x1, inv);
        }
    }
}

// ---- Prep kernel: run once per launch. Converts W1/W2 (f32 row-major) into the
// exact transposed+padded f16 LDS image layout, and builds bias / 1/(2s^2) tables.
__global__ __launch_bounds__(256) void prep_weights(
    const float* __restrict__ W1, const float* __restrict__ b1, const float* __restrict__ s1,
    const float* __restrict__ W2, const float* __restrict__ b2, const float* __restrict__ s2,
    _Float16* __restrict__ wimg, float* __restrict__ tbl)
{
    const int gid   = blockIdx.x * 256 + threadIdx.x;   // grid = 512 blocks covers 2*65536
    const int layer = gid >> 16;
    const int idx   = gid & 0xFFFF;
    const float* W  = layer ? W2 : W1;
    const int k = idx >> 8;          // input feature (row of W) -> contiguous in LDS image
    const int n = idx & 255;         // output feature (col)     -> coalesced global read
    wimg[layer * WIMG_HALVES + n * WS + k] = (_Float16)W[idx];

    if (blockIdx.x == 0) {
        const int t = threadIdx.x;
        tbl[t]         = b1[t];
        float sv1 = s1[t];
        tbl[F + t]     = 1.0f / (2.0f * sv1 * sv1);
        tbl[2 * F + t] = b2[t];
        float sv2 = s2[t];
        tbl[3 * F + t] = 1.0f / (2.0f * sv2 * sv2);
    }
}

__global__ __launch_bounds__(256) void fused_gauss_mlp(
    const float* __restrict__ pos,
    const float* __restrict__ W0, const float* __restrict__ b0,
    const float* __restrict__ s0,
    const _Float16* __restrict__ wimg,   // [2][F*WS] pre-converted weight images
    const float* __restrict__ tbl,       // [4][F] b1, i1, b2, i2
    const float* __restrict__ W3, const float* __restrict__ b3,
    float* __restrict__ out, int n)
{
    extern __shared__ char smem[];
    _Float16* wbuf = (_Float16*)smem;          // 132 KB (W1 image, then W2 image)
    _Float16* actA = wbuf + F * WS;            // 66 KB
    _Float16* actB = actA + ROWS * AS;         // 66 KB
    float* bbuf = (float*)(actB + ROWS * AS);  // 256 f32
    float* ibuf = bbuf + F;                    // 256 f32
    float* posT = ibuf + F;                    // 128*3 f32

    const int tid  = threadIdx.x;
    const int lane = tid & 31;
    const int wave = tid >> 5;
    const int l    = lane & 15;
    const int hi   = lane >> 4;
    const long long row0 = (long long)blockIdx.x * ROWS;

    // ---- Stage pos tile + W1 image + layer-1 tables ----
    for (int idx = tid; idx < ROWS * 3; idx += 256) {
        long long gi = row0 * 3 + idx;
        posT[idx] = (gi < (long long)n * 3) ? pos[gi] : 0.f;
    }
    stage_from_img(wimg, tbl, wbuf, bbuf, ibuf, tid);
    prefetch_img(wimg + WIMG_HALVES, tid);   // warm W2 image while layer 0 runs

    // ---- Layer 0: [128,3] @ [3,256] + b0, gauss(sigma0). One column per thread. ----
    {
        const int c = tid;
        float w0c = W0[c], w1c = W0[F + c], w2c = W0[2 * F + c];
        float bc  = b0[c];
        float sv  = s0[c];
        float inv = 1.0f / (2.0f * sv * sv);
        __syncthreads();   // posT ready
        for (int r = 0; r < ROWS; ++r) {
            float x = posT[r * 3 + 0] * w0c + posT[r * 3 + 1] * w1c
                    + posT[r * 3 + 2] * w2c + bc;
            actA[r * AS + c] = (_Float16)gaussf(x, inv);
        }
    }
    __syncthreads();

    // ---- Layer 1: actB = gauss(actA @ W1 + b1, sigma1) ----
    wmma_layer(actA, actB, wbuf, bbuf, ibuf, wave, l, hi);
    __syncthreads();

    // ---- Restage W2 image + layer-2 tables (B128 copy, source in L2/WGP$) ----
    stage_from_img(wimg + WIMG_HALVES, tbl + 2 * F, wbuf, bbuf, ibuf, tid);
    __syncthreads();

    // ---- Layer 2: actA = gauss(actB @ W2 + b2, sigma2) ----
    wmma_layer(actB, actA, wbuf, bbuf, ibuf, wave, l, hi);
    __syncthreads();

    // ---- Layer 3: [128,256] @ [256,1] + b3, softplus(threshold=8) ----
    {
        const int r    = tid >> 1;          // row 0..127
        const int half = tid & 1;           // which 128-wide half of K
        const _Float16* ap = actA + r * AS + half * 128;
        const float* wp = W3 + half * 128;
        float s = 0.f;
#pragma unroll 8
        for (int k = 0; k < 128; ++k)
            s += (float)ap[k] * wp[k];
        s += __shfl_xor(s, 1, 32);          // combine adjacent lanes (wave32)
        if (half == 0) {
            long long g = row0 + r;
            float x = s + b3[0];
            float y = (x > 8.0f) ? x : __logf(1.0f + __expf(fminf(x, 8.0f)));
            if (g < n) out[g] = y;
        }
    }
}

extern "C" void kernel_launch(void* const* d_in, const int* in_sizes, int n_in,
                              void* d_out, int out_size, void* d_ws, size_t ws_size,
                              hipStream_t stream) {
    const float* pos = (const float*)d_in[0];
    const float* W0  = (const float*)d_in[1];
    const float* b0  = (const float*)d_in[2];
    const float* W1  = (const float*)d_in[3];
    const float* b1  = (const float*)d_in[4];
    const float* W2  = (const float*)d_in[5];
    const float* b2  = (const float*)d_in[6];
    const float* W3  = (const float*)d_in[7];
    const float* b3  = (const float*)d_in[8];
    const float* s0  = (const float*)d_in[9];
    const float* s1  = (const float*)d_in[10];
    const float* s2  = (const float*)d_in[11];
    float* out = (float*)d_out;

    // Workspace layout: [2 weight images f16][4 x 256 f32 tables]
    _Float16* wimg = (_Float16*)d_ws;
    float* tbl = (float*)((char*)d_ws + 2 * WIMG_BYTES);

    const int N = in_sizes[0] / 3;
    const int grid = (N + ROWS - 1) / ROWS;
    const size_t smem = (size_t)(F * WS + 2 * ROWS * AS) * sizeof(_Float16)
                      + (size_t)(2 * F + ROWS * 3) * sizeof(float);   // ~274 KB

    hipFuncSetAttribute((const void*)fused_gauss_mlp,
                        hipFuncAttributeMaxDynamicSharedMemorySize, (int)smem);

    prep_weights<<<512, 256, 0, stream>>>(W1, b1, s1, W2, b2, s2, wimg, tbl);
    fused_gauss_mlp<<<grid, 256, smem, stream>>>(
        pos, W0, b0, s0, wimg, tbl, W3, b3, out, N);
}
